// TCM_GNN_13735305412915
// MI455X (gfx1250) — matrix operand order repaired and checked by reading, not compiled, not run on hardware
//
#include <hip/hip_runtime.h>
#include <hip/hip_bf16.h>
#include <math.h>

// ---------------------------------------------------------------------------
// CDNA5 / gfx1250 implementation of the 2-layer relational GAT forward.
// All dense matmuls run on v_wmma_f32_16x16x32_bf16 (bf16 in, f32 accum),
// with B pre-swizzled into fragment order and A loaded as b128 vectors.
// Irregular edge softmax / scatter uses f32 VALU + global atomics.
// ---------------------------------------------------------------------------

typedef __bf16 bf16_t;
typedef __attribute__((ext_vector_type(16))) __bf16 v16bf;
typedef __attribute__((ext_vector_type(8)))  float  v8f;

#define NNODES 20000
#define NGRAPH 20
#define NGPAD  32      // NGRAPH padded to a full 16x16 WMMA M-tile multiple
#define NREL   4
#define NEDGE  160000
#define DDIM   64
#define NHEAD  4
#define NIN    56

__device__ __forceinline__ float gelu_exact(float x) {
    return 0.5f * x * (1.0f + erff(x * 0.70710678118654752f));
}
__device__ __forceinline__ float lrelu02(float x) { return x > 0.f ? x : 0.2f * x; }

// order-preserving float<->uint map for atomicMax-based segment max
__device__ __forceinline__ unsigned fenc(float f) {
    unsigned u = __float_as_uint(f);
    return (u & 0x80000000u) ? ~u : (u | 0x80000000u);
}
__device__ __forceinline__ float fdec(unsigned k) {
    unsigned u = (k & 0x80000000u) ? (k & 0x7fffffffu) : ~k;
    return __uint_as_float(u);
}

// ---------------------------------------------------------------------------
// Generic fill
// ---------------------------------------------------------------------------
__global__ void fill_u32_kernel(unsigned* __restrict__ p, unsigned v, long n) {
    long i = (long)blockIdx.x * blockDim.x + threadIdx.x;
    long stride = (long)gridDim.x * blockDim.x;
    for (; i < n; i += stride) p[i] = v;
}

// ---------------------------------------------------------------------------
// Pack f32 [Ksrc x C] weights into bf16 WMMA-B fragment order (K zero-padded
// to Kdst, a multiple of 32).  Layout: for k-block kb, column n, the 32
// fragment values live contiguously at dst[(kb*C + n)*32 + half*16 + e],
// holding W[kb*32 + 8*half + (e&7) + 16*(e>>3)][n]  (ISA §7.12.2 dense 32x16 16-bit B:
// lanes 0-15 carry K=0..7,16..23, lanes 16-31 carry K=8..15,24..31).
// A lane then loads its whole fragment as one contiguous 32-byte chunk.
// ---------------------------------------------------------------------------
__global__ void pack_w_frag_kernel(const float* __restrict__ src, bf16_t* __restrict__ dst,
                                   int Ksrc, int Kdst, int C) {
    int i = blockIdx.x * blockDim.x + threadIdx.x;
    int tot = (Kdst >> 5) * C * 32;
    if (i >= tot) return;
    int t = i & 31;
    int rem = i >> 5;            // kb*C + n
    int n = rem % C;
    int kb = rem / C;
    int k = (kb << 5) + ((t & 16) >> 1) + (t & 7) + ((t & 8) << 1);       // t = half*16 + e
    dst[i] = (k < Ksrc) ? (bf16_t)src[(size_t)k * C + n] : (bf16_t)0.f;
}

// feat_all (N x 56) -> featA (N x 64, zero padded)
__global__ void pad_feat_kernel(const float* __restrict__ fa, float* __restrict__ out, int n) {
    int i = blockIdx.x * blockDim.x + threadIdx.x;
    if (i >= n * DDIM) return;
    int node = i >> 6, c = i & 63;
    out[i] = (c < NIN) ? fa[(size_t)node * NIN + c] : 0.f;
}

// ---------------------------------------------------------------------------
// WMMA GEMM:  Y[nrows x C] = X[nrows x K](f32 -> bf16) @ W[K x C](bf16 frags)
// Requirements (guaranteed by the host side): nrows % 16 == 0, K % 64 == 0,
// C % 16 == 0, X/W/Y 256B aligned.  One wave owns one 16x16 tile; k-loop is
// unrolled by 64 (two chained WMMAs / iter).  No per-lane control flow, so
// EXEC is all-1s at every WMMA as the ISA requires.
// A fragment (16x32 bf16, ISA §7.12.2): lane half h in {0,1} starts at K=8h;
// elems 0..7 -> K = 8h+e, elems 8..15 -> K = 8h+16+(e-8): two 8-float runs,
// loaded as 4x float4.
// ---------------------------------------------------------------------------
__global__ void gemm_bf16_wmma(const float* __restrict__ X, int ldx,
                               const bf16_t* __restrict__ Wf,
                               const float* __restrict__ bias,
                               float* __restrict__ Y, int ldy,
                               int nrows, int K, int C, int do_gelu) {
    const int wave    = blockIdx.x * (blockDim.x >> 5) + (threadIdx.x >> 5);
    const int lane    = threadIdx.x & 31;
    const int tiles_x = C >> 4;
    const int tile_y  = wave / tiles_x;
    const int tile_x  = wave - tile_y * tiles_x;
    const int m0      = tile_y << 4;
    if (m0 >= nrows) return;                 // wave-uniform: EXEC stays all-1s
    const int half = lane >> 4;
    const int l16  = lane & 15;
    const int ncol = (tile_x << 4) + l16;
    const int kA   = half << 3;              // A half-offset within a 32-block
    const float*  xrow  = X + (size_t)(m0 + l16) * ldx;
    const bf16_t* bbase = Wf + ((size_t)ncol << 5) + (half << 4);

    v8f acc = {0.f, 0.f, 0.f, 0.f, 0.f, 0.f, 0.f, 0.f};

    for (int k0 = 0; k0 < K; k0 += 64) {
        __builtin_prefetch(xrow + k0 + 64, 0, 3);   // speculative, unguarded
        // ---- A fragments: 8x b128 loads, pack to bf16 ----
        const float4 x0 = *(const float4*)(xrow + k0 + kA);
        const float4 x1 = *(const float4*)(xrow + k0 + kA + 4);
        const float4 x2 = *(const float4*)(xrow + k0 + kA + 16);
        const float4 x3 = *(const float4*)(xrow + k0 + kA + 20);
        const float4 x4 = *(const float4*)(xrow + k0 + kA + 32);
        const float4 x5 = *(const float4*)(xrow + k0 + kA + 36);
        const float4 x6 = *(const float4*)(xrow + k0 + kA + 48);
        const float4 x7 = *(const float4*)(xrow + k0 + kA + 52);
        v16bf a0, a1;
        a0[0]  = (bf16_t)x0.x; a0[1]  = (bf16_t)x0.y; a0[2]  = (bf16_t)x0.z; a0[3]  = (bf16_t)x0.w;
        a0[4]  = (bf16_t)x1.x; a0[5]  = (bf16_t)x1.y; a0[6]  = (bf16_t)x1.z; a0[7]  = (bf16_t)x1.w;
        a0[8]  = (bf16_t)x2.x; a0[9]  = (bf16_t)x2.y; a0[10] = (bf16_t)x2.z; a0[11] = (bf16_t)x2.w;
        a0[12] = (bf16_t)x3.x; a0[13] = (bf16_t)x3.y; a0[14] = (bf16_t)x3.z; a0[15] = (bf16_t)x3.w;
        a1[0]  = (bf16_t)x4.x; a1[1]  = (bf16_t)x4.y; a1[2]  = (bf16_t)x4.z; a1[3]  = (bf16_t)x4.w;
        a1[4]  = (bf16_t)x5.x; a1[5]  = (bf16_t)x5.y; a1[6]  = (bf16_t)x5.z; a1[7]  = (bf16_t)x5.w;
        a1[8]  = (bf16_t)x6.x; a1[9]  = (bf16_t)x6.y; a1[10] = (bf16_t)x6.z; a1[11] = (bf16_t)x6.w;
        a1[12] = (bf16_t)x7.x; a1[13] = (bf16_t)x7.y; a1[14] = (bf16_t)x7.z; a1[15] = (bf16_t)x7.w;
        // ---- B fragments: contiguous 32B per lane (2x b128 each) ----
        const int kb0 = k0 >> 5;
        const v16bf b0 = *(const v16bf*)(bbase + ((size_t)kb0 * C << 5));
        const v16bf b1 = *(const v16bf*)(bbase + ((size_t)(kb0 + 1) * C << 5));
        acc = __builtin_amdgcn_wmma_f32_16x16x32_bf16(false, a0, false, b0,
                                                      (short)0, acc, false, false);
        acc = __builtin_amdgcn_wmma_f32_16x16x32_bf16(false, a1, false, b1,
                                                      (short)0, acc, false, false);
    }

    const float bv = bias ? bias[ncol] : 0.f;
#pragma unroll
    for (int r2 = 0; r2 < 8; ++r2) {
        float vv = acc[r2] + bv;
        if (do_gelu) vv = gelu_exact(vv);
        Y[(size_t)(m0 + (half << 3) + r2) * ldy + ncol] = vv;
    }
}

// ---------------------------------------------------------------------------
// GAT node-side attention logits:  el/er[n,h] = sum_f ft[n,h,f]*al/ar[h,f]
// ---------------------------------------------------------------------------
__global__ void act_node_kernel(const float* __restrict__ ft,
                                const float* __restrict__ al, const float* __restrict__ ar,
                                float* __restrict__ el, float* __restrict__ er, int n) {
    int i = blockIdx.x * blockDim.x + threadIdx.x;
    if (i >= n * NHEAD) return;
    int node = i >> 2, h = i & 3;
    const float* f = ft + (size_t)node * 16 + h * 4;
    float sl = 0.f, sr = 0.f;
    for (int c = 0; c < 4; ++c) { sl += f[c] * al[h * 4 + c]; sr += f[c] * ar[h * 4 + c]; }
    el[i] = sl; er[i] = sr;
}

__global__ void env_node_kernel(const float* __restrict__ ft,
                                const float* __restrict__ al, const float* __restrict__ ar,
                                float* __restrict__ el, float* __restrict__ er, int n) {
    int i = blockIdx.x * blockDim.x + threadIdx.x;
    if (i >= n * NHEAD) return;
    int node = i >> 2, h = i & 3;
    const float* f = ft + (size_t)node * 256 + h * 64;
    float sl = 0.f, sr = 0.f;
    for (int c = 0; c < 64; ++c) { sl += f[c] * al[h * 64 + c]; sr += f[c] * ar[h * 64 + c]; }
    el[i] = sl; er[i] = sr;
}

// ---------------------------------------------------------------------------
// Edge softmax passes (segmented by destination v), thread per (edge, head)
// ---------------------------------------------------------------------------
__global__ void edge_max_kernel(const int* __restrict__ u, const int* __restrict__ v,
                                const float* __restrict__ el, const float* __restrict__ er,
                                unsigned* __restrict__ mkey, int ne) {
    int i = blockIdx.x * blockDim.x + threadIdx.x;
    if (i >= ne * NHEAD) return;
    int e = i >> 2, h = i & 3;
    float val = lrelu02(el[u[e] * NHEAD + h] + er[v[e] * NHEAD + h]);
    atomicMax(&mkey[v[e] * NHEAD + h], fenc(val));
}

__global__ void edge_sum_kernel(const int* __restrict__ u, const int* __restrict__ v,
                                const float* __restrict__ el, const float* __restrict__ er,
                                const unsigned* __restrict__ mkey,
                                float* __restrict__ exb, float* __restrict__ ssum, int ne) {
    int i = blockIdx.x * blockDim.x + threadIdx.x;
    if (i >= ne * NHEAD) return;
    int e = i >> 2, h = i & 3;
    float val = lrelu02(el[u[e] * NHEAD + h] + er[v[e] * NHEAD + h]);
    float ex  = expf(val - fdec(mkey[v[e] * NHEAD + h]));
    exb[i] = ex;
    atomicAdd(&ssum[v[e] * NHEAD + h], ex);
}

// a = ex/s (optionally * gate[v]*gate[u]);  rst[v,h,:] += a * ft[u,h,:]
__global__ void edge_aggr_kernel(const int* __restrict__ u, const int* __restrict__ v,
                                 const float* __restrict__ exb, const float* __restrict__ ssum,
                                 const float* __restrict__ ft, float* __restrict__ rst, int F,
                                 const float* __restrict__ inb, const float* __restrict__ outb,
                                 int ne) {
    int i = blockIdx.x * blockDim.x + threadIdx.x;
    if (i >= ne * NHEAD) return;
    int e = i >> 2, h = i & 3;
    int vi = v[e], ui = u[e];
    float a = exb[i] / ssum[vi * NHEAD + h];
    if (inb) a *= inb[vi] * outb[ui];                 // gumbel ST gate (0/1)
    const float* fu = ft + (size_t)ui * (NHEAD * F) + h * F;
    float*       rv = rst + (size_t)vi * (NHEAD * F) + h * F;
    for (int c = 0; c < F; ++c) atomicAdd(&rv[c], a * fu[c]);
}

// logits = gelu(rst + b).mean(heads); gumbel straight-through forward == one-hot
__global__ void act_post_kernel(const float* __restrict__ rst, const float* __restrict__ actb,
                                const float* __restrict__ gin, const float* __restrict__ gout,
                                float* __restrict__ inb, float* __restrict__ outb, int n) {
    int i = blockIdx.x * blockDim.x + threadIdx.x;
    if (i >= n) return;
    float lg[4];
    for (int c = 0; c < 4; ++c) {
        float s = 0.f;
        for (int h = 0; h < NHEAD; ++h)
            s += gelu_exact(rst[(size_t)i * 16 + h * 4 + c] + actb[h * 4 + c]);
        lg[c] = s * 0.25f;
    }
    // argmax(softmax((l+g)/tau)) == argmax(l+g); first index wins ties
    inb[i]  = (lg[0] + gin[(size_t)i * 2 + 0] >= lg[1] + gin[(size_t)i * 2 + 1]) ? 1.f : 0.f;
    outb[i] = (lg[2] + gout[(size_t)i * 2 + 0] >= lg[3] + gout[(size_t)i * 2 + 1]) ? 1.f : 0.f;
}

// hs[r,n,c] = sum_h gelu(rst + res + env_b)
__global__ void env_post_kernel(const float* __restrict__ rst, const float* __restrict__ res,
                                const float* __restrict__ envb, float* __restrict__ hsout, int n) {
    int i = blockIdx.x * blockDim.x + threadIdx.x;
    if (i >= n * DDIM) return;
    int node = i >> 6, c = i & 63;
    float acc = 0.f;
    for (int h = 0; h < NHEAD; ++h) {
        int o = h * DDIM + c;
        acc += gelu_exact(rst[(size_t)node * 256 + o] + res[(size_t)node * 256 + o] + envb[o]);
    }
    hsout[i] = acc;
}

// layernorm over last dim of hs[R,n,64]; write transposed hn[n, r*64+c]
__global__ void ln_kernel(const float* __restrict__ hs, const float* __restrict__ g,
                          const float* __restrict__ b, float* __restrict__ hn, int n) {
    int i = blockIdx.x * blockDim.x + threadIdx.x;
    if (i >= NREL * n) return;
    int r = i / n, node = i - r * n;
    const float* row = hs + (size_t)i * DDIM;
    float mu = 0.f;
    for (int c = 0; c < DDIM; ++c) mu += row[c];
    mu *= (1.f / DDIM);
    float var = 0.f;
    for (int c = 0; c < DDIM; ++c) { float d = row[c] - mu; var += d * d; }
    var *= (1.f / DDIM);
    float inv = rsqrtf(var + 1e-5f);
    float* out = hn + (size_t)node * 256 + r * DDIM;
    for (int c = 0; c < DDIM; ++c) out[c] = (row[c] - mu) * inv * g[c] + b[c];
}

// per-node 4x4 attention over q/k/v rows laid out as (n*4) x 64
__global__ void attn_kernel(const float* __restrict__ q, const float* __restrict__ k,
                            const float* __restrict__ v, float* __restrict__ out, int n) {
    int i = blockIdx.x * blockDim.x + threadIdx.x;
    if (i >= n * NREL) return;
    int node = i >> 2, r = i & 3;
    const float* qr = q + (size_t)i * DDIM;
    float sc[4], mx = -1e30f;
    for (int s2 = 0; s2 < 4; ++s2) {
        const float* ks = k + ((size_t)node * 4 + s2) * DDIM;
        float d = 0.f;
        for (int c = 0; c < DDIM; ++c) d += qr[c] * ks[c];
        sc[s2] = d * 0.125f;                          // 1/sqrt(64)
        mx = fmaxf(mx, sc[s2]);
    }
    float se = 0.f;
    for (int s2 = 0; s2 < 4; ++s2) { sc[s2] = expf(sc[s2] - mx); se += sc[s2]; }
    float inv = 1.f / se;
    float* o = out + (size_t)node * 256 + r * DDIM;
    for (int c = 0; c < DDIM; ++c) {
        float acc = 0.f;
        for (int s2 = 0; s2 < 4; ++s2) acc += sc[s2] * v[((size_t)node * 4 + s2) * DDIM + c];
        o[c] = acc * inv;
    }
}

// ---------------------------------------------------------------------------
// Graph pooling + head
// ---------------------------------------------------------------------------
__global__ void pool_accum_kernel(const float* __restrict__ feat, const float* __restrict__ pool,
                                  const int* __restrict__ ngr, float* __restrict__ rosum,
                                  float* __restrict__ cnt, int n) {
    int i = blockIdx.x * blockDim.x + threadIdx.x;
    if (i >= n) return;
    int g = ngr[i];
    atomicAdd(&cnt[g], 1.f);
    float pm = pool[i];
    if (pm != 0.f) {
        const float* f = feat + (size_t)i * DDIM;
        for (int c = 0; c < DDIM; ++c) atomicAdd(&rosum[g * DDIM + c], f[c] * pm);
    }
}

__global__ void pool_final_kernel(const float* __restrict__ rosum, const float* __restrict__ cnt,
                                  float* __restrict__ ro, float* __restrict__ dout, int g64) {
    int i = blockIdx.x * blockDim.x + threadIdx.x;
    if (i >= g64) return;
    float v = rosum[i] / cnt[i >> 6];
    ro[i] = v;                  // rows 20..31 of ro stay zero (padded M-tile)
    dout[NGRAPH + i] = v;       // ro output follows pred in the flat tuple
}

// LDS tree-reduction head:  pred[g] = b2 + sum_j hmid[g, j] * w2[j]
__global__ void pred_kernel(const float* __restrict__ hmid, const float* __restrict__ w2,
                            const float* __restrict__ b2, float* __restrict__ dout) {
    __shared__ float sm[32];
    const int g = blockIdx.x;
    const int j = threadIdx.x;
    sm[j] = hmid[g * 32 + j] * w2[j];
    __syncthreads();
    for (int s = 16; s > 0; s >>= 1) {
        if (j < s) sm[j] += sm[j + s];
        __syncthreads();
    }
    if (j == 0) dout[g] = sm[0] + b2[0];
}

// ---------------------------------------------------------------------------
// Host side
// ---------------------------------------------------------------------------
extern "C" void kernel_launch(void* const* d_in, const int* in_sizes, int n_in,
                              void* d_out, int out_size, void* d_ws, size_t ws_size,
                              hipStream_t stream) {
    (void)in_sizes; (void)n_in; (void)out_size; (void)ws_size;
    const int N = NNODES;

    // ---- input mapping (setup_inputs dict order; params tree flattened with
    //      layers list first, each layer dict in sorted-key order) ----
    const float* feat_all = (const float*)d_in[0];
    enum { P_ACT_AL = 0, P_ACT_AR, P_ACT_B, P_ACT_FC, P_BK, P_BP, P_BQ, P_BV,
           P_ENV_AL, P_ENV_AR, P_ENV_B, P_ENV_FC, P_ENV_RES, P_LN_B, P_LN_G,
           P_WK, P_WP, P_WQ, P_WV };
    auto LP = [&](int l, int p) -> const float* { return (const float*)d_in[1 + l * 19 + p]; };
    const float* reg_b1   = (const float*)d_in[39];
    const float* reg_b2   = (const float*)d_in[40];
    const float* reg_w1   = (const float*)d_in[41];
    const float* reg_w2   = (const float*)d_in[42];
    const float* pool_ids = (const float*)d_in[43];
    const int*   edges_u  = (const int*)d_in[44];
    const int*   edges_v  = (const int*)d_in[45];
    const int*   node_gr  = (const int*)d_in[46];
    const float* gin_all  = (const float*)d_in[47];
    const float* gout_all = (const float*)d_in[48];

    // ---- workspace bump allocator (256B aligned blocks) ----
    char* base = (char*)d_ws;
    size_t off = 0;
    auto alloc = [&](size_t bytes) -> char* {
        char* p = base + off;
        off += (bytes + 255) & ~(size_t)255;
        return p;
    };
    // bf16 fragment-packed weights (K padded to mult of 32)
    bf16_t *w_act[2], *w_envfc[2], *w_envres[2], *w_q[2], *w_k[2], *w_v[2], *w_p[2];
    for (int l = 0; l < 2; ++l) {
        w_act[l]    = (bf16_t*)alloc(64 * 16 * 2);
        w_envfc[l]  = (bf16_t*)alloc(64 * 256 * 2);
        w_envres[l] = (bf16_t*)alloc(64 * 256 * 2);
        w_q[l]      = (bf16_t*)alloc(64 * 64 * 2);
        w_k[l]      = (bf16_t*)alloc(64 * 64 * 2);
        w_v[l]      = (bf16_t*)alloc(64 * 64 * 2);
        w_p[l]      = (bf16_t*)alloc(256 * 64 * 2);
    }
    bf16_t* w_r1 = (bf16_t*)alloc(64 * 32 * 2);
    // f32 intermediates
    float* featA   = (float*)alloc((size_t)N * 64 * 4);
    float* featB   = (float*)alloc((size_t)N * 64 * 4);
    float* ft_act  = (float*)alloc((size_t)N * 16 * 4);
    float* el_     = (float*)alloc((size_t)N * NHEAD * 4);
    float* er_     = (float*)alloc((size_t)N * NHEAD * 4);
    unsigned* mkey = (unsigned*)alloc((size_t)N * NHEAD * 4);
    float* ssum    = (float*)alloc((size_t)N * NHEAD * 4);
    float* rst_act = (float*)alloc((size_t)N * 16 * 4);
    float* inb     = (float*)alloc((size_t)N * 4);
    float* outb    = (float*)alloc((size_t)N * 4);
    float* exb     = (float*)alloc((size_t)NEDGE * NHEAD * 4);
    float* ft_env  = (float*)alloc((size_t)N * 256 * 4);
    float* res_env = (float*)alloc((size_t)N * 256 * 4);
    float* rst_env = (float*)alloc((size_t)N * 256 * 4);
    float* hs      = (float*)alloc((size_t)NREL * N * 64 * 4);
    float* hn      = (float*)alloc((size_t)N * 256 * 4);
    float* qb      = (float*)alloc((size_t)N * 256 * 4);
    float* kb      = (float*)alloc((size_t)N * 256 * 4);
    float* vb      = (float*)alloc((size_t)N * 256 * 4);
    float* attn    = (float*)alloc((size_t)N * 256 * 4);
    float* rosum   = (float*)alloc((size_t)NGRAPH * 64 * 4);
    float* cnt     = (float*)alloc((size_t)NGRAPH * 4);
    float* ro      = (float*)alloc((size_t)NGPAD * 64 * 4);   // padded M-tile
    float* hmid    = (float*)alloc((size_t)NGPAD * 32 * 4);

    auto fillu = [&](void* p, unsigned v, size_t n) {
        int blocks = (int)((n + 255) / 256);
        if (blocks > 4096) blocks = 4096;
        fill_u32_kernel<<<blocks, 256, 0, stream>>>((unsigned*)p, v, (long)n);
    };
    auto pack = [&](const float* src, bf16_t* dst, int Ks, int Kd, int C) {
        int tot = (Kd >> 5) * C * 32;
        pack_w_frag_kernel<<<(tot + 255) / 256, 256, 0, stream>>>(src, dst, Ks, Kd, C);
    };
    auto gemm = [&](const float* X, int ldx, const bf16_t* W, const float* bias,
                    float* Y, int ldy, int nrows, int K, int C, int gl) {
        int tiles = (nrows / 16) * (C / 16);     // nrows % 16 == 0 guaranteed
        gemm_bf16_wmma<<<(tiles + 7) / 8, 256, 0, stream>>>(X, ldx, W, bias, Y, ldy,
                                                            nrows, K, C, gl);
    };

    // ---- pack weights into fragment order (pad K: 56 -> 64) ----
    const int ind[2] = {NIN, DDIM};
    for (int l = 0; l < 2; ++l) {
        pack(LP(l, P_ACT_FC),  w_act[l],    ind[l], 64, 16);
        pack(LP(l, P_ENV_FC),  w_envfc[l],  ind[l], 64, 256);
        pack(LP(l, P_ENV_RES), w_envres[l], ind[l], 64, 256);
        pack(LP(l, P_WQ), w_q[l], 64, 64, 64);
        pack(LP(l, P_WK), w_k[l], 64, 64, 64);
        pack(LP(l, P_WV), w_v[l], 64, 64, 64);
        pack(LP(l, P_WP), w_p[l], 256, 256, 64);
    }
    pack(reg_w1, w_r1, 64, 64, 32);

    // ---- input features, padded to stride 64 ----
    pad_feat_kernel<<<(N * 64 + 255) / 256, 256, 0, stream>>>(feat_all, featA, N);

    const unsigned NEGINF_KEY = 0x007FFFFFu;   // fenc(-inf)
    const int ETH = NEDGE * NHEAD;
    float* fin = featA;
    float* fout = featB;

    for (int l = 0; l < 2; ++l) {
        for (int r = 0; r < NREL; ++r) {
            const int* u = edges_u + (size_t)r * NEDGE;
            const int* v = edges_v + (size_t)r * NEDGE;
            const float* gin  = gin_all  + ((size_t)(l * NREL + r) * N) * 2;
            const float* gout = gout_all + ((size_t)(l * NREL + r) * N) * 2;

            // ---- act GAT (gate logits), h=4 f=4 ----
            gemm(fin, 64, w_act[l], nullptr, ft_act, 16, N, 64, 16, 0);
            fillu(mkey, NEGINF_KEY, (size_t)N * NHEAD);
            fillu(ssum, 0u, (size_t)N * NHEAD);
            fillu(rst_act, 0u, (size_t)N * 16);
            act_node_kernel<<<(N * NHEAD + 255) / 256, 256, 0, stream>>>(
                ft_act, LP(l, P_ACT_AL), LP(l, P_ACT_AR), el_, er_, N);
            edge_max_kernel<<<(ETH + 255) / 256, 256, 0, stream>>>(u, v, el_, er_, mkey, NEDGE);
            edge_sum_kernel<<<(ETH + 255) / 256, 256, 0, stream>>>(u, v, el_, er_, mkey,
                                                                   exb, ssum, NEDGE);
            edge_aggr_kernel<<<(ETH + 255) / 256, 256, 0, stream>>>(
                u, v, exb, ssum, ft_act, rst_act, 4, nullptr, nullptr, NEDGE);
            act_post_kernel<<<(N + 255) / 256, 256, 0, stream>>>(
                rst_act, LP(l, P_ACT_B), gin, gout, inb, outb, N);

            // ---- env GAT, h=4 f=64 ----
            gemm(fin, 64, w_envfc[l],  nullptr, ft_env,  256, N, 64, 256, 0);
            gemm(fin, 64, w_envres[l], nullptr, res_env, 256, N, 64, 256, 0);
            fillu(mkey, NEGINF_KEY, (size_t)N * NHEAD);
            fillu(ssum, 0u, (size_t)N * NHEAD);
            fillu(rst_env, 0u, (size_t)N * 256);
            env_node_kernel<<<(N * NHEAD + 255) / 256, 256, 0, stream>>>(
                ft_env, LP(l, P_ENV_AL), LP(l, P_ENV_AR), el_, er_, N);
            edge_max_kernel<<<(ETH + 255) / 256, 256, 0, stream>>>(u, v, el_, er_, mkey, NEDGE);
            edge_sum_kernel<<<(ETH + 255) / 256, 256, 0, stream>>>(u, v, el_, er_, mkey,
                                                                   exb, ssum, NEDGE);
            edge_aggr_kernel<<<(ETH + 255) / 256, 256, 0, stream>>>(
                u, v, exb, ssum, ft_env, rst_env, 64, inb, outb, NEDGE);
            env_post_kernel<<<(N * DDIM + 255) / 256, 256, 0, stream>>>(
                rst_env, res_env, LP(l, P_ENV_B), hs + (size_t)r * N * DDIM, N);
        }

        // ---- layernorm + per-node 4-way attention + output projection ----
        ln_kernel<<<(NREL * N + 255) / 256, 256, 0, stream>>>(hs, LP(l, P_LN_G),
                                                              LP(l, P_LN_B), hn, N);
        gemm(hn, 64, w_q[l], LP(l, P_BQ), qb, 64, N * 4, 64, 64, 0);
        gemm(hn, 64, w_k[l], LP(l, P_BK), kb, 64, N * 4, 64, 64, 0);
        gemm(hn, 64, w_v[l], LP(l, P_BV), vb, 64, N * 4, 64, 64, 0);
        attn_kernel<<<(N * 4 + 255) / 256, 256, 0, stream>>>(qb, kb, vb, attn, N);
        gemm(attn, 256, w_p[l], LP(l, P_BP), fout, 64, N, 256, 64, 0);

        float* t = fin; fin = fout; fout = t;
    }

    // ---- pooled readout + regression head ----
    fillu(rosum, 0u, (size_t)NGRAPH * 64);
    fillu(cnt, 0u, (size_t)NGRAPH);
    fillu(ro, 0u, (size_t)NGPAD * 64);     // zero the padded rows 20..31
    pool_accum_kernel<<<(N + 255) / 256, 256, 0, stream>>>(fin, pool_ids, node_gr,
                                                           rosum, cnt, N);
    pool_final_kernel<<<(NGRAPH * 64 + 255) / 256, 256, 0, stream>>>(
        rosum, cnt, ro, (float*)d_out, NGRAPH * 64);
    gemm(ro, 64, w_r1, reg_b1, hmid, 32, NGPAD, 64, 32, 1);   // gelu fused
    pred_kernel<<<NGRAPH, 32, 0, stream>>>(hmid, reg_w2, reg_b2, (float*)d_out);
}
